// LSTMWithAttention_16853451670181
// MI455X (gfx1250) — matrix operand order repaired and checked by reading, not compiled, not run on hardware
//
#include <hip/hip_runtime.h>
#include <hip/hip_bf16.h>
#include <math.h>
#include <stdint.h>

typedef __attribute__((ext_vector_type(16))) __bf16 v16bf;
typedef __attribute__((ext_vector_type(8)))  float  v8f;

#define DEVINL __device__ __forceinline__

static constexpr int Bdim = 8, Tdim = 2048, Hdim = 1024, D1 = 256, OUTd = 256;
static constexpr int BT = Bdim * Tdim;         // 16384
static constexpr int NREC_BLOCKS = 32;         // LSTM recurrence blocks

// ---------- CDNA5 async global->LDS path (guarded; falls back to sync copy) ----------
#if defined(__HIP_DEVICE_COMPILE__)
#if __has_builtin(__builtin_amdgcn_global_load_async_to_lds_b128) && \
    __has_builtin(__builtin_amdgcn_s_wait_asynccnt)
#define USE_ASYNC_LDS 1
#endif
#if __has_builtin(__builtin_amdgcn_tensor_load_to_lds) && \
    __has_builtin(__builtin_amdgcn_s_wait_tensorcnt)
#define USE_TDM 1
#endif
#endif

#ifdef USE_ASYNC_LDS
typedef int async_v4i __attribute__((__vector_size__(16)));
typedef __attribute__((address_space(1))) async_v4i* async_gptr;
typedef __attribute__((address_space(3))) async_v4i* async_lptr;
#endif

DEVINL void copy16_g2l(void* lds_dst, const void* gsrc) {
#ifdef USE_ASYNC_LDS
  // per-lane: LDS[dst] = MEM[src], tracked by ASYNCcnt (global_load_async_to_lds_b128)
  __builtin_amdgcn_global_load_async_to_lds_b128(
      (async_gptr)(uintptr_t)gsrc,
      (async_lptr)(unsigned)(uintptr_t)lds_dst,
      0, 0);
#else
  *reinterpret_cast<uint4*>(lds_dst) = *reinterpret_cast<const uint4*>(gsrc);
#endif
}
DEVINL void wait_async() {
#ifdef USE_ASYNC_LDS
  __builtin_amdgcn_s_wait_asynccnt(0);
#endif
}

#ifdef USE_TDM
typedef unsigned int tdm_v4u __attribute__((__vector_size__(16)));
typedef int tdm_v8i __attribute__((__vector_size__(32)));
typedef int tdm_v4i __attribute__((__vector_size__(16)));

// TDM: load a 2D tile of bf16 (rows x cols, row stride in elements) global -> LDS.
// D# packing per CDNA5 ISA 8.3/8.4 (group0: count/lds/global/type; group1: dims).
DEVINL void tdm_load_2d_bf16(void* lds_dst, const void* gsrc,
                             int rows, int cols, int row_stride_elems) {
  unsigned ldsoff = (unsigned)(uintptr_t)lds_dst;              // LDS byte offset
  unsigned long long ga = (unsigned long long)(uintptr_t)gsrc; // global byte addr
  tdm_v4u g0;
  g0[0] = 1u;                                        // count=1 valid descriptor
  g0[1] = ldsoff;                                    // lds_addr [63:32]
  g0[2] = (unsigned)(ga & 0xffffffffu);              // global_addr[31:0] -> bits 95:64
  g0[3] = (unsigned)((ga >> 32) & 0x01ffffffu)       // global_addr[56:32]
        | 0x80000000u;                               // type=2 ("image") bits 127:126
  tdm_v8i g1;
  g1[0] = 0x00010000;                                // wg_mask=0, data_size=1 (2B)
  g1[1] = (int)((unsigned)cols << 16);               // tensor_dim0[15:0] @ bits 63:48
  g1[2] = (int)(((unsigned)cols >> 16) & 0xffffu)    // tensor_dim0[31:16]
        | (int)((unsigned)rows << 16);               // tensor_dim1[15:0] @ bits 95:80
  g1[3] = (int)(((unsigned)rows >> 16) & 0xffffu)    // tensor_dim1[31:16]
        | (int)((unsigned)cols << 16);               // tile_dim0 @ bits 127:112
  g1[4] = rows;                                      // tile_dim1 (tile_dim2=0)
  g1[5] = row_stride_elems;                          // tensor_dim0_stride[31:0]
  g1[6] = 0;                                         // stride hi / dim1 stride
  g1[7] = 0;
  tdm_v4i z4 = {0, 0, 0, 0};
#if __has_include(<hip/amd_detail/amd_gfx1250_TDM.h>)
  tdm_v8i z8 = {0, 0, 0, 0, 0, 0, 0, 0};             // 6-arg toolchain form
  __builtin_amdgcn_tensor_load_to_lds(g0, g1, z4, z4, z8, 0);
#else
  __builtin_amdgcn_tensor_load_to_lds(g0, g1, z4, z4, 0);  // ROCm 7.2 5-arg form
#endif
}
DEVINL void wait_tensor() { __builtin_amdgcn_s_wait_tensorcnt(0); }
#endif

// ---------- helpers ----------
DEVINL __bf16 f2bf(float f) {
  union { float f; unsigned u; } v; v.f = f;
  unsigned r = v.u + 0x7fffu + ((v.u >> 16) & 1u);   // round-to-nearest-even
  unsigned short h = (unsigned short)(r >> 16);
  union { unsigned short s; __bf16 b; } o; o.s = h; return o.b;
}
DEVINL float geluf(float x) { return 0.5f * x * (1.0f + erff(x * 0.70710678118654752f)); }
DEVINL float sigm(float x)  { return 1.0f / (1.0f + expf(-x)); }
DEVINL v8f vzero8() { v8f z; for (int i = 0; i < 8; ++i) z[i] = 0.0f; return z; }

DEVINL void grid_sync(unsigned* bar, unsigned nb) {
  __syncthreads();
  if (threadIdx.x == 0) {
    __threadfence();
    unsigned gen = __hip_atomic_load(bar + 1, __ATOMIC_ACQUIRE, __HIP_MEMORY_SCOPE_AGENT);
    unsigned v   = __hip_atomic_fetch_add(bar, 1u, __ATOMIC_ACQ_REL, __HIP_MEMORY_SCOPE_AGENT);
    if (v == nb - 1u) {
      __hip_atomic_store(bar, 0u, __ATOMIC_RELAXED, __HIP_MEMORY_SCOPE_AGENT);
      __hip_atomic_fetch_add(bar + 1, 1u, __ATOMIC_RELEASE, __HIP_MEMORY_SCOPE_AGENT);
    } else {
      while (__hip_atomic_load(bar + 1, __ATOMIC_ACQUIRE, __HIP_MEMORY_SCOPE_AGENT) == gen)
        __builtin_amdgcn_s_sleep(1);
    }
  }
  __syncthreads();
}

// ---------- elementwise kernels ----------
__global__ void k_cvt_bf16(const float* __restrict__ s, __bf16* __restrict__ d, int n) {
  int i = blockIdx.x * 256 + threadIdx.x;
  if (i < n) d[i] = f2bf(s[i]);
}
__global__ void k_add(const float* a, const float* b, float* d, int n) {
  int i = blockIdx.x * 256 + threadIdx.x;
  if (i < n) d[i] = a[i] + b[i];
}
__global__ void k_zero_f32(float* d, int n) {
  int i = blockIdx.x * 256 + threadIdx.x;
  if (i < n) d[i] = 0.0f;
}
// x2 blend with next_mask/next_token, GELU, write bf16 into concat buffer second half
__global__ void k_blend(const float* __restrict__ x2, const float* __restrict__ mask,
                        const float* __restrict__ ntok, __bf16* __restrict__ cat) {
  size_t i = (size_t)blockIdx.x * 256 + threadIdx.x;
  if (i >= (size_t)BT * Hdim) return;
  size_t r = i / Hdim; int hcol = (int)(i % Hdim);
  float mk = mask[r];
  float v  = x2[i] * (1.0f - mk) + mk * ntok[hcol];
  cat[r * (size_t)(2 * Hdim) + Hdim + hcol] = f2bf(geluf(v));
}

// ---------- generic WMMA GEMM:  out = act(A[M,K] @ W[N,K]^T + bias) ----------
// BM=128, BN=64, K-chunks of 32, 256 threads = 8 waves, double-buffered async staging
__global__ __launch_bounds__(256) void k_gemm(const __bf16* __restrict__ A, int lda,
                                              const __bf16* __restrict__ W, int ldw,
                                              const float* __restrict__ bias,
                                              float* __restrict__ outF, __bf16* __restrict__ outB,
                                              int ldc, int M, int N, int K, int act) {
  __shared__ __bf16 As[2][128 * 32];
  __shared__ __bf16 Ws[2][64 * 32];
  const int m0 = blockIdx.y * 128, n0 = blockIdx.x * 64;
  const int tid = threadIdx.x, wave = tid >> 5, lane = tid & 31;

  v8f acc[4];
  for (int j = 0; j < 4; ++j) acc[j] = vzero8();

  // hoisted per-thread staging addresses (row = tid>>2, 16B segment = tid&3)
  const int srow = tid >> 2, sseg = (tid & 3) * 8;
  const __bf16* Ap0 = A + (size_t)(m0 + srow) * lda + sseg;
  const __bf16* Ap1 = Ap0 + (size_t)64 * lda;
  const __bf16* Wp  = W + (size_t)(n0 + srow) * ldw + sseg;
  __bf16* Al0 = &As[0][srow * 32 + sseg];
  __bf16* Wl0 = &Ws[0][srow * 32 + sseg];

  auto stage = [&](int buf, int k0) {
    copy16_g2l(Al0 + buf * (128 * 32),            Ap0 + k0);
    copy16_g2l(Al0 + buf * (128 * 32) + 64 * 32,  Ap1 + k0);
    copy16_g2l(Wl0 + buf * (64 * 32),             Wp  + k0);
  };

  stage(0, 0);
  int cur = 0;
  for (int k0 = 0; k0 < K; k0 += 32) {
    wait_async();          // my async copies into As[cur]/Ws[cur] done
    __syncthreads();       // everyone's done; previous buffer free for overwrite
    if (k0 + 32 < K) stage(cur ^ 1, k0 + 32);

    // A fragment (16x32): lane group g: elems 0-15 <-> K 16g..16g+15 contiguous
    union AV { v16bf v; uint4 q[2]; } af;
    {
      int row = wave * 16 + (lane & 15);
      int g = lane >> 4;
      af.q[0] = *reinterpret_cast<const uint4*>(&As[cur][row * 32 + 16 * g]);
      af.q[1] = *reinterpret_cast<const uint4*>(&As[cur][row * 32 + 16 * g + 8]);
    }
    #pragma unroll
    for (int jt = 0; jt < 4; ++jt) {
      // B fragment (32x16): lane n = lane&15, K range 16*(lane>>4)..+16 contiguous
      union BV { v16bf v; uint4 q[2]; } bv;
      int nrow = jt * 16 + (lane & 15);
      int kb = 16 * (lane >> 4);
      bv.q[0] = *reinterpret_cast<const uint4*>(&Ws[cur][nrow * 32 + kb]);
      bv.q[1] = *reinterpret_cast<const uint4*>(&Ws[cur][nrow * 32 + kb + 8]);
      acc[jt] = __builtin_amdgcn_wmma_f32_16x16x32_bf16(false, af.v, false, bv.v,
                                                        (short)0, acc[jt], false, false);
    }
    cur ^= 1;
  }

  // epilogue: C layout m = r + 8*(lane>=16), n = lane&15
  const int mb = m0 + wave * 16;
  #pragma unroll
  for (int jt = 0; jt < 4; ++jt) {
    #pragma unroll
    for (int r = 0; r < 8; ++r) {
      int m = mb + r + 8 * (lane >> 4);
      int n = n0 + jt * 16 + (lane & 15);
      if (m < M && n < N) {
        float v = acc[jt][r];
        if (bias) v += bias[n];
        if (act == 1) v = geluf(v);
        size_t o = (size_t)m * ldc + n;
        if (outF) outF[o] = v;
        if (outB) outB[o] = f2bf(v);
      }
    }
  }
}

// ---------- LSTM recurrence ----------
// 32 blocks; block bx owns gate columns [bx*32, bx*32+32) of each of i,f,g,o.
// whh slice (128x1024 bf16, 256KB) TDM-loaded resident in LDS; h_prev converted to a
// 16x1024 bf16 A-panel in LDS each step. Per step: G[16,128] = Hpanel @ Wslice^T via
// WMMA, LSTM cell math, grid barrier across the 32 blocks.
__global__ __launch_bounds__(256) void k_lstm(const __bf16* __restrict__ whh_bf,
                                              const float* __restrict__ xg,
                                              float* __restrict__ hbuf0, float* __restrict__ hbuf1,
                                              float* __restrict__ rnn_out,
                                              unsigned* __restrict__ bar) {
  extern __shared__ char smem[];
  __bf16* Wl = (__bf16*)smem;                                  // [128][1024] 256KB
  float*  Gl = (float*)(smem + 128 * 1024 * 2);                // [16][128]   8KB
  __bf16* Ab = (__bf16*)(smem + 128 * 1024 * 2 + 16 * 128 * 4); // [16][1024] 32KB
  const int tid = threadIdx.x, wave = tid >> 5, lane = tid & 31;
  const int n0 = blockIdx.x * 32;

  // stage whh slice: 4 gate tiles of 32 rows x 1024 cols (row = gate*H + n0 + r)
#ifdef USE_TDM
  if (wave == 0) {
    #pragma unroll
    for (int gate = 0; gate < 4; ++gate)
      tdm_load_2d_bf16(&Wl[gate * 32 * 1024],
                       whh_bf + (size_t)(gate * Hdim + n0) * Hdim,
                       32, 1024, 1024);
  }
#else
  for (int idx = tid; idx < 128 * 128; idx += 256) {
    int rr = idx >> 7, seg = idx & 127;
    int gate = rr >> 5, col = n0 + (rr & 31);
    copy16_g2l(&Wl[rr * 1024 + seg * 8],
               whh_bf + (size_t)(gate * Hdim + col) * Hdim + seg * 8);
  }
#endif
  // zero A-panel padding rows 8..15 (never touched again)
  {
    const uint4 z = make_uint4(0, 0, 0, 0);
    for (int idx = tid; idx < 1024; idx += 256) {
      int row = 8 + (idx >> 7), seg = idx & 127;
      *reinterpret_cast<uint4*>(&Ab[row * 1024 + seg * 8]) = z;
    }
  }
#ifdef USE_TDM
  if (wave == 0) wait_tensor();
#else
  wait_async();
#endif
  __syncthreads();

  const int mb = tid >> 5;   // batch for pointwise (0..7)
  const int j  = tid & 31;   // column within slice
  float c = 0.0f;

  for (int tt = 0; tt < Tdim; ++tt) {
    const float* hp = (tt & 1) ? hbuf1 : hbuf0;
    float*       hn = (tt & 1) ? hbuf0 : hbuf1;

    // convert h_prev (8x1024 f32) into bf16 A-panel rows 0..7
    for (int idx = tid; idx < 1024; idx += 256) {
      int row = idx >> 7, seg = idx & 127;
      const float* src = hp + row * Hdim + seg * 8;
      union { uint4 q; __bf16 h[8]; } u;
      #pragma unroll
      for (int e = 0; e < 8; ++e) u.h[e] = f2bf(src[e]);
      *reinterpret_cast<uint4*>(&Ab[row * 1024 + seg * 8]) = u.q;
    }
    __syncthreads();

    v8f acc = vzero8();
    for (int k0 = 0; k0 < Hdim; k0 += 32) {
      union AV { v16bf v; uint4 q[2]; } af;
      int arow = lane & 15, g = lane >> 4;
      af.q[0] = *reinterpret_cast<const uint4*>(&Ab[arow * 1024 + k0 + 16 * g]);
      af.q[1] = *reinterpret_cast<const uint4*>(&Ab[arow * 1024 + k0 + 16 * g + 8]);
      union BV { v16bf v; uint4 q[2]; } bv;
      int nrow = wave * 16 + (lane & 15);
      int kb = k0 + 16 * (lane >> 4);
      bv.q[0] = *reinterpret_cast<const uint4*>(&Wl[nrow * 1024 + kb]);
      bv.q[1] = *reinterpret_cast<const uint4*>(&Wl[nrow * 1024 + kb + 8]);
      acc = __builtin_amdgcn_wmma_f32_16x16x32_bf16(false, af.v, false, bv.v,
                                                    (short)0, acc, false, false);
    }
    #pragma unroll
    for (int r = 0; r < 8; ++r) {
      int m = r + 8 * (lane >> 4);
      int n = wave * 16 + (lane & 15);
      Gl[m * 128 + n] = acc[r];
    }
    __syncthreads();

    {
      const size_t xb = ((size_t)mb * Tdim + tt) * (4 * Hdim);
      float gi = Gl[mb * 128 + j]      + xg[xb + 0 * Hdim + n0 + j];
      float gf = Gl[mb * 128 + 32 + j] + xg[xb + 1 * Hdim + n0 + j];
      float gg = Gl[mb * 128 + 64 + j] + xg[xb + 2 * Hdim + n0 + j];
      float go = Gl[mb * 128 + 96 + j] + xg[xb + 3 * Hdim + n0 + j];
      float i_ = sigm(gi), f_ = sigm(gf), o_ = sigm(go), g_ = tanhf(gg);
      c = f_ * c + i_ * g_;
      float h = o_ * tanhf(c);
      hn[mb * Hdim + n0 + j] = h;
      rnn_out[((size_t)mb * Tdim + tt) * Hdim + n0 + j] = h;
    }
    grid_sync(bar, NREC_BLOCKS);
  }
}

// ---------- LayerNorm (writes bf16 to h_bf and concat[:, :H]) ----------
__global__ __launch_bounds__(256) void k_layernorm(const float* __restrict__ x,
                                                   const float* __restrict__ gw,
                                                   const float* __restrict__ bw,
                                                   __bf16* __restrict__ hbf,
                                                   __bf16* __restrict__ cat) {
  __shared__ float red[256];
  int r = blockIdx.x;
  const float* row = x + (size_t)r * Hdim;
  float v[4], s = 0.0f;
  for (int i = 0; i < 4; ++i) { v[i] = row[threadIdx.x + 256 * i]; s += v[i]; }
  red[threadIdx.x] = s; __syncthreads();
  for (int st = 128; st > 0; st >>= 1) { if (threadIdx.x < st) red[threadIdx.x] += red[threadIdx.x + st]; __syncthreads(); }
  float mu = red[0] / (float)Hdim; __syncthreads();
  float s2 = 0.0f;
  for (int i = 0; i < 4; ++i) { float d = v[i] - mu; s2 += d * d; }
  red[threadIdx.x] = s2; __syncthreads();
  for (int st = 128; st > 0; st >>= 1) { if (threadIdx.x < st) red[threadIdx.x] += red[threadIdx.x + st]; __syncthreads(); }
  float rstd = rsqrtf(red[0] / (float)Hdim + 1e-5f);
  for (int i = 0; i < 4; ++i) {
    int hc = threadIdx.x + 256 * i;
    __bf16 b = f2bf((v[i] - mu) * rstd * gw[hc] + bw[hc]);
    hbf[(size_t)r * Hdim + hc] = b;
    cat[(size_t)r * (2 * Hdim) + hc] = b;
  }
}

// ---------- banded attention (MEM=64), one 16-query tile per block ----------
__global__ __launch_bounds__(128) void k_attn(const __bf16* __restrict__ hbf,
                                              const __bf16* __restrict__ Kbf,
                                              __bf16* __restrict__ cat) {
  extern __shared__ char smem[];
  __bf16* Hw = (__bf16*)smem;                                 // [96][1024] value window
  float*  S  = (float*)(smem + 96 * 1024 * 2);                // [16][80]
  __bf16* P  = (__bf16*)(smem + 96 * 1024 * 2 + 16 * 80 * 4); // [16][96]
  const int qt = blockIdx.x, b = blockIdx.y;
  const int q0 = qt * 16;
  const int lo = (q0 - 63 > 0) ? q0 - 63 : 0;
  const int L  = q0 + 16 - lo;   // valid window rows (<= 79)
  const size_t rb = (size_t)b * Tdim;
  const int tid = threadIdx.x, wave = tid >> 5, lane = tid & 31;

  // stage value window rows [lo, lo+L): one TDM tile (or per-lane async); pad to 96
  {
    const uint4 z = make_uint4(0, 0, 0, 0);
#ifdef USE_TDM
    if (wave == 0) tdm_load_2d_bf16(Hw, hbf + (rb + lo) * Hdim, L, 1024, 1024);
    for (int idx = tid; idx < 96 * 128; idx += 128) {
      int kk = idx >> 7, seg = idx & 127;
      if (kk >= L) *reinterpret_cast<uint4*>(&Hw[kk * Hdim + seg * 8]) = z;
    }
#else
    for (int idx = tid; idx < 96 * 128; idx += 128) {
      int kk = idx >> 7, seg = idx & 127;
      if (kk < L) copy16_g2l(&Hw[kk * Hdim + seg * 8],
                             hbf + (rb + lo + kk) * Hdim + seg * 8);
      else        *reinterpret_cast<uint4*>(&Hw[kk * Hdim + seg * 8]) = z;
    }
#endif
  }

  // scores S[16,80] = Q @ K^T / 32, fragments straight from global (overlaps staging)
  const int g = lane >> 4;
  const __bf16* qptr = hbf + (rb + q0 + (lane & 15)) * Hdim;
  for (int jt = wave; jt < 5; jt += 4) {
    int keyrow = lo + jt * 16 + (lane & 15);
    if (keyrow > Tdim - 1) keyrow = Tdim - 1;   // clamped rows are masked below
    const __bf16* kptr = Kbf + (rb + keyrow) * Hdim;
    v8f acc = vzero8();
    for (int k0 = 0; k0 < Hdim; k0 += 32) {
      union AV { v16bf v; uint4 q[2]; } af;
      af.q[0] = *reinterpret_cast<const uint4*>(qptr + k0 + 16 * g);
      af.q[1] = *reinterpret_cast<const uint4*>(qptr + k0 + 16 * g + 8);
      union BV { v16bf v; uint4 q[2]; } bv;
      bv.q[0] = *reinterpret_cast<const uint4*>(kptr + k0 + 16 * g);
      bv.q[1] = *reinterpret_cast<const uint4*>(kptr + k0 + 16 * g + 8);
      acc = __builtin_amdgcn_wmma_f32_16x16x32_bf16(false, af.v, false, bv.v,
                                                    (short)0, acc, false, false);
    }
    #pragma unroll
    for (int r = 0; r < 8; ++r) {
      int m = r + 8 * (lane >> 4);
      int n = jt * 16 + (lane & 15);
      if (n < 80) {
        int q = q0 + m, kx = lo + n;
        S[m * 80 + n] = (kx <= q && kx >= q - 63) ? acc[r] * 0.03125f : -INFINITY;
      }
    }
  }
#ifdef USE_TDM
  if (wave == 0) wait_tensor();   // value window resident before it is consumed
#else
  wait_async();
#endif
  __syncthreads();

  // softmax per row (row q always has >=1 valid key -> no NaN)
  if (tid < 16) {
    int m = tid;
    float mx = -INFINITY;
    for (int n = 0; n < 80; ++n) mx = fmaxf(mx, S[m * 80 + n]);
    float sum = 0.0f;
    for (int n = 0; n < 80; ++n) { float e = expf(S[m * 80 + n] - mx); S[m * 80 + n] = e; sum += e; }
    float inv = 1.0f / sum;
    for (int n = 0; n < 80; ++n) P[m * 96 + n] = f2bf(S[m * 80 + n] * inv);
    for (int n = 80; n < 96; ++n) P[m * 96 + n] = f2bf(0.0f);
  }
  __syncthreads();

  // context C[16,1024] = P[16,96] @ Hw[96,1024], write bf16 into concat[:, H:2H]
  for (int nt = wave; nt < 64; nt += 4) {
    v8f acc = vzero8();
    for (int ks = 0; ks < 3; ++ks) {
      union AV { v16bf v; uint4 q[2]; } af;
      int mrow = lane & 15;
      af.q[0] = *reinterpret_cast<const uint4*>(P + mrow * 96 + ks * 32 + 16 * g);
      af.q[1] = *reinterpret_cast<const uint4*>(P + mrow * 96 + ks * 32 + 16 * g + 8);
      union BV { v16bf v; __bf16 e[16]; } bv;
      int hcol = nt * 16 + (lane & 15);
      int kb = ks * 32 + 16 * (lane >> 4);
      #pragma unroll
      for (int e = 0; e < 16; ++e) bv.e[e] = Hw[(kb + e) * Hdim + hcol];
      acc = __builtin_amdgcn_wmma_f32_16x16x32_bf16(false, af.v, false, bv.v,
                                                    (short)0, acc, false, false);
    }
    #pragma unroll
    for (int r = 0; r < 8; ++r) {
      int m = r + 8 * (lane >> 4);
      int hcol = nt * 16 + (lane & 15);
      cat[(rb + q0 + m) * (size_t)(2 * Hdim) + Hdim + hcol] = f2bf(acc[r]);
    }
  }
}

// ---------- cosine detector ----------
__global__ __launch_bounds__(256) void k_cosine(const float* __restrict__ comb,
                                                const float* __restrict__ ntok,
                                                float* __restrict__ out) {
  __shared__ float rd[256], rn[256];
  int r = blockIdx.x;
  const float* row = comb + (size_t)r * Hdim;
  float dot = 0.0f, nrm = 0.0f;
  for (int i = threadIdx.x; i < Hdim; i += 256) { float v = row[i]; dot += v * ntok[i]; nrm += v * v; }
  rd[threadIdx.x] = dot; rn[threadIdx.x] = nrm; __syncthreads();
  for (int st = 128; st > 0; st >>= 1) {
    if (threadIdx.x < st) { rd[threadIdx.x] += rd[threadIdx.x + st]; rn[threadIdx.x] += rn[threadIdx.x + st]; }
    __syncthreads();
  }
  if (threadIdx.x == 0) {
    float nn = 0.0f;
    for (int i = 0; i < Hdim; ++i) { float v = ntok[i]; nn += v * v; }
    float den = fmaxf(sqrtf(rn[0]), 1e-8f) * fmaxf(sqrtf(nn), 1e-8f);
    out[r] = rd[0] / den;
  }
}

// ---------- host ----------
extern "C" void kernel_launch(void* const* d_in, const int* in_sizes, int n_in,
                              void* d_out, int out_size, void* d_ws, size_t ws_size,
                              hipStream_t stream) {
  (void)in_sizes; (void)n_in; (void)out_size; (void)ws_size;
  const float* in1   = (const float*)d_in[0];
  const float* in2   = (const float*)d_in[1];
  const float* nmask = (const float*)d_in[2];
  const float* e1w1 = (const float*)d_in[3];  const float* e1b1 = (const float*)d_in[4];
  const float* e1w2 = (const float*)d_in[5];  const float* e1b2 = (const float*)d_in[6];
  const float* e2w1 = (const float*)d_in[7];  const float* e2b1 = (const float*)d_in[8];
  const float* e2w2 = (const float*)d_in[9];  const float* e2b2 = (const float*)d_in[10];
  const float* mw   = (const float*)d_in[11]; const float* mb   = (const float*)d_in[12];
  const float* wih  = (const float*)d_in[13]; const float* whh  = (const float*)d_in[14];
  const float* bih  = (const float*)d_in[15]; const float* bhh  = (const float*)d_in[16];
  const float* lng  = (const float*)d_in[17]; const float* lnb  = (const float*)d_in[18];
  const float* aw   = (const float*)d_in[19];
  const float* cw1  = (const float*)d_in[20]; const float* cb1  = (const float*)d_in[21];
  const float* cw2  = (const float*)d_in[22]; const float* cb2  = (const float*)d_in[23];
  const float* ntok = (const float*)d_in[24];
  const float* hw1  = (const float*)d_in[25]; const float* hb1  = (const float*)d_in[26];
  const float* hw2  = (const float*)d_in[27]; const float* hb2  = (const float*)d_in[28];

  char* ws = (char*)d_ws;
  size_t off = 0;
  auto alloc = [&](size_t bytes) -> void* { void* p = ws + off; off += (bytes + 255) & ~(size_t)255; return p; };

  __bf16* in1_bf  = (__bf16*)alloc((size_t)BT * D1 * 2);
  __bf16* in2_bf  = (__bf16*)alloc((size_t)BT * D1 * 2);
  __bf16* w_e1w1  = (__bf16*)alloc((size_t)Hdim * D1 * 2);
  __bf16* w_e1w2  = (__bf16*)alloc((size_t)Hdim * Hdim * 2);
  __bf16* w_e2w1  = (__bf16*)alloc((size_t)Hdim * D1 * 2);
  __bf16* w_e2w2  = (__bf16*)alloc((size_t)Hdim * Hdim * 2);
  __bf16* w_merge = (__bf16*)alloc((size_t)Hdim * 2 * Hdim * 2);
  __bf16* w_wih   = (__bf16*)alloc((size_t)4 * Hdim * Hdim * 2);
  __bf16* w_whh   = (__bf16*)alloc((size_t)4 * Hdim * Hdim * 2);
  __bf16* w_attn  = (__bf16*)alloc((size_t)Hdim * Hdim * 2);
  __bf16* w_cw1   = (__bf16*)alloc((size_t)Hdim * 2 * Hdim * 2);
  __bf16* w_cw2   = (__bf16*)alloc((size_t)Hdim * Hdim * 2);
  __bf16* w_hw1   = (__bf16*)alloc((size_t)Hdim * Hdim * 2);
  __bf16* w_hw2   = (__bf16*)alloc((size_t)OUTd * Hdim * 2);
  float*  bias_sum = (float*)alloc(4 * Hdim * 4);
  __bf16* h1_bf   = (__bf16*)alloc((size_t)BT * Hdim * 2);
  __bf16* h2_bf   = (__bf16*)alloc((size_t)BT * Hdim * 2);
  __bf16* merged  = (__bf16*)alloc((size_t)BT * 2 * Hdim * 2);   // concat(GELU(x1), GELU(x2m))
  float*  x2_f    = (float*)alloc((size_t)BT * Hdim * 4);
  __bf16* x_bf    = (__bf16*)alloc((size_t)BT * Hdim * 2);
  float*  xg_f    = (float*)alloc((size_t)BT * 4 * Hdim * 4);
  float*  hbuf    = (float*)alloc((size_t)(2 * Bdim * Hdim + 8) * 4); // ping/pong + barrier
  float*  rnn_out = (float*)alloc((size_t)BT * Hdim * 4);
  __bf16* h_bf    = (__bf16*)alloc((size_t)BT * Hdim * 2);
  __bf16* K_bf    = (__bf16*)alloc((size_t)BT * Hdim * 2);
  __bf16* cat_bf  = (__bf16*)alloc((size_t)BT * 2 * Hdim * 2);   // concat(h_ln, context)
  __bf16* c1_bf   = (__bf16*)alloc((size_t)BT * Hdim * 2);
  float*  comb_f  = (float*)alloc((size_t)BT * Hdim * 4);
  __bf16* comb_bf = (__bf16*)alloc((size_t)BT * Hdim * 2);
  __bf16* hh_bf   = (__bf16*)alloc((size_t)BT * Hdim * 2);

  auto cvt = [&](const float* s, __bf16* d, size_t n) {
    k_cvt_bf16<<<(unsigned)((n + 255) / 256), 256, 0, stream>>>(s, d, (int)n);
  };
  cvt(in1, in1_bf, (size_t)BT * D1);
  cvt(in2, in2_bf, (size_t)BT * D1);
  cvt(e1w1, w_e1w1, (size_t)Hdim * D1);
  cvt(e1w2, w_e1w2, (size_t)Hdim * Hdim);
  cvt(e2w1, w_e2w1, (size_t)Hdim * D1);
  cvt(e2w2, w_e2w2, (size_t)Hdim * Hdim);
  cvt(mw,   w_merge, (size_t)Hdim * 2 * Hdim);
  cvt(wih,  w_wih, (size_t)4 * Hdim * Hdim);
  cvt(whh,  w_whh, (size_t)4 * Hdim * Hdim);
  cvt(aw,   w_attn, (size_t)Hdim * Hdim);
  cvt(cw1,  w_cw1, (size_t)Hdim * 2 * Hdim);
  cvt(cw2,  w_cw2, (size_t)Hdim * Hdim);
  cvt(hw1,  w_hw1, (size_t)Hdim * Hdim);
  cvt(hw2,  w_hw2, (size_t)OUTd * Hdim);
  k_add<<<16, 256, 0, stream>>>(bih, bhh, bias_sum, 4 * Hdim);

  auto gemm = [&](const __bf16* A, int lda, const __bf16* W, int ldw, const float* bias,
                  float* oF, __bf16* oB, int ldc, int N, int K, int act) {
    dim3 g(N / 64, BT / 128);
    k_gemm<<<g, 256, 0, stream>>>(A, lda, W, ldw, bias, oF, oB, ldc, BT, N, K, act);
  };

  // embeddings -> merged concat (GELU baked in)
  gemm(in1_bf, D1, w_e1w1, D1, e1b1, nullptr, h1_bf, Hdim, Hdim, D1, 1);
  gemm(h1_bf, Hdim, w_e1w2, Hdim, e1b2, nullptr, merged, 2 * Hdim, Hdim, Hdim, 1);
  gemm(in2_bf, D1, w_e2w1, D1, e2b1, nullptr, h2_bf, Hdim, Hdim, D1, 1);
  gemm(h2_bf, Hdim, w_e2w2, Hdim, e2b2, x2_f, nullptr, Hdim, Hdim, Hdim, 0);
  k_blend<<<(unsigned)((size_t)BT * Hdim / 256), 256, 0, stream>>>(x2_f, nmask, ntok, merged);
  // merge + LSTM input projection
  gemm(merged, 2 * Hdim, w_merge, 2 * Hdim, mb, nullptr, x_bf, Hdim, Hdim, 2 * Hdim, 0);
  gemm(x_bf, Hdim, w_wih, Hdim, bias_sum, xg_f, nullptr, 4 * Hdim, 4 * Hdim, Hdim, 0);

  // recurrence (h0/c0 = 0; barrier state re-zeroed every launch)
  k_zero_f32<<<(2 * Bdim * Hdim + 8 + 255) / 256, 256, 0, stream>>>(hbuf, 2 * Bdim * Hdim + 8);
  unsigned* bar = (unsigned*)(hbuf + 2 * Bdim * Hdim);
  size_t lstm_lds = (size_t)128 * 1024 * 2 + (size_t)16 * 128 * 4 + (size_t)16 * 1024 * 2;
  k_lstm<<<NREC_BLOCKS, 256, lstm_lds, stream>>>(w_whh, xg_f, hbuf, hbuf + Bdim * Hdim,
                                                 rnn_out, bar);

  // layernorm -> h_bf and concat[:, :H]
  k_layernorm<<<BT, 256, 0, stream>>>(rnn_out, lng, lnb, h_bf, cat_bf);
  // attention K projection (no bias)
  gemm(h_bf, Hdim, w_attn, Hdim, nullptr, nullptr, K_bf, Hdim, Hdim, Hdim, 0);
  // banded attention -> concat[:, H:2H]
  size_t attn_lds = (size_t)96 * 1024 * 2 + (size_t)16 * 80 * 4 + (size_t)16 * 96 * 2;
  k_attn<<<dim3(Tdim / 16, Bdim), 128, attn_lds, stream>>>(h_bf, K_bf, cat_bf);

  // combiner
  gemm(cat_bf, 2 * Hdim, w_cw1, 2 * Hdim, cb1, nullptr, c1_bf, Hdim, Hdim, 2 * Hdim, 1);
  gemm(c1_bf, Hdim, w_cw2, Hdim, cb2, comb_f, comb_bf, Hdim, Hdim, Hdim, 0);

  // cosine detector -> d_out[BT*OUT ..)
  float* outF = (float*)d_out;
  k_cosine<<<BT, 256, 0, stream>>>(comb_f, ntok, outF + (size_t)BT * OUTd);

  // head -> pred into d_out[0 ..)
  gemm(comb_bf, Hdim, w_hw1, Hdim, hb1, nullptr, hh_bf, Hdim, Hdim, Hdim, 1);
  gemm(hh_bf, Hdim, w_hw2, Hdim, hb2, outF, nullptr, OUTd, OUTd, Hdim, 0);
}
